// VanillaMultiHeadAttention_88562225643742
// MI455X (gfx1250) — compile-verified, matrix operand order checked by reading, not run on hardware
//
#include <hip/hip_runtime.h>
#include <hip/hip_bf16.h>
#include <stdint.h>

// ---------------------------------------------------------------- types
typedef __attribute__((ext_vector_type(16))) __bf16  v16bf;
typedef __attribute__((ext_vector_type(8)))  __bf16  v8bf;
typedef __attribute__((ext_vector_type(8)))  float   v8f;

// problem constants
static constexpr int Bc = 4, Sc = 2048, Dc = 1024, Hc = 16, DKc = 64;
static constexpr int BSc = Bc * Sc;                 // 8192 rows
static constexpr int HDc = Hc * DKc;                // 1024

// ---------------------------------------------------------------- helpers
// bf16 fragment from bf16 memory: lane row base + K offset 8*half; chunks at
// +0 (K = 8h..8h+7) and +16 (K = 8h+16..8h+23) per the 16-bit A/B VGPR layout.
__device__ __forceinline__ v16bf load_frag_bf16(const __bf16* p) {
  v8bf c0 = *reinterpret_cast<const v8bf*>(p);
  v8bf c1 = *reinterpret_cast<const v8bf*>(p + 16);
  return __builtin_shufflevector(c0, c1, 0,1,2,3,4,5,6,7,8,9,10,11,12,13,14,15);
}

// bf16 fragment from fp32 memory (native packed v_cvt_pk_bf16_f32)
__device__ __forceinline__ v16bf load_frag_f32(const float* p) {
  v8f x0 = *reinterpret_cast<const v8f*>(p);
  v8f x1 = *reinterpret_cast<const v8f*>(p + 16);
  v8bf b0 = __builtin_convertvector(x0, v8bf);
  v8bf b1 = __builtin_convertvector(x1, v8bf);
  return __builtin_shufflevector(b0, b1, 0,1,2,3,4,5,6,7,8,9,10,11,12,13,14,15);
}

__device__ __forceinline__ v8f wmma_bf16(v16bf a, v16bf b, v8f c) {
  return __builtin_amdgcn_wmma_f32_16x16x32_bf16(false, a, false, b,
                                                 (short)0, c, false, false);
}

// ---------------------------------------------------------------- kernels
// fp32 W[K=1024][N=1024] -> bf16 Wt[N][K]
__global__ void transpose_w_kernel(const float* __restrict__ W,
                                   __bf16* __restrict__ Wt) {
  int idx = blockIdx.x * blockDim.x + threadIdx.x;     // over 1M
  int n = idx >> 10, k = idx & 1023;
  Wt[idx] = (__bf16)W[(size_t)k * 1024 + n];
}

// fp32 -> bf16 bulk convert, 8 elements/thread
__global__ void cvt_bf16_kernel(const float* __restrict__ src,
                                __bf16* __restrict__ dst) {
  int i = blockIdx.x * blockDim.x + threadIdx.x;
  v8f x = *reinterpret_cast<const v8f*>(src + (size_t)i * 8);
  *reinterpret_cast<v8bf*>(dst + (size_t)i * 8) = __builtin_convertvector(x, v8bf);
}

// C[M=8192,N=1024] = A(bf16)[M,K=1024] @ W[K,N], B supplied as bf16 Wt[N][K].
// 2x2 register-blocked: one 32x32 tile per wave (4 WMMAs per 4 fragment
// loads -> 2 b128 loads per WMMA). mode: 0 = bf16 store (scaled),
// 2 = V store transposed per head: Vt[((b*H+h)*64+nn)*S + s].
__global__ __launch_bounds__(256)
void proj_gemm_kernel(const __bf16* __restrict__ A,
                      const __bf16* __restrict__ Wt,
                      __bf16* __restrict__ Cout,
                      float scale, int mode) {
  constexpr int M = BSc, N = HDc, K = Dc;
  int wave = blockIdx.x * 8 + (threadIdx.x >> 5);
  int lane = threadIdx.x & 31;
  int half = lane >> 4, l16 = lane & 15;
  int mt = wave >> 5;          // / (N/32 = 32)
  int nt = wave & 31;
  if (mt >= (M >> 5)) return;

  const __bf16* arow0 = A  + (size_t)(mt * 32 + l16) * K + 8 * half;
  const __bf16* arow1 = arow0 + (size_t)16 * K;
  const __bf16* brow0 = Wt + (size_t)(nt * 32 + l16) * K + 8 * half;
  const __bf16* brow1 = brow0 + (size_t)16 * K;

  v8f acc00 = {}, acc01 = {}, acc10 = {}, acc11 = {};
  for (int k0 = 0; k0 < K; k0 += 32) {
    __builtin_prefetch(arow0 + k0 + 512, 0, 1);
    __builtin_prefetch(brow0 + k0 + 512, 0, 1);
    v16bf a0 = load_frag_bf16(arow0 + k0);
    v16bf a1 = load_frag_bf16(arow1 + k0);
    v16bf b0 = load_frag_bf16(brow0 + k0);
    v16bf b1 = load_frag_bf16(brow1 + k0);
    acc00 = wmma_bf16(a0, b0, acc00);
    acc01 = wmma_bf16(a0, b1, acc01);
    acc10 = wmma_bf16(a1, b0, acc10);
    acc11 = wmma_bf16(a1, b1, acc11);
  }

#pragma unroll
  for (int sm = 0; sm < 2; ++sm) {
#pragma unroll
    for (int sn = 0; sn < 2; ++sn) {
      const v8f& acc = sm == 0 ? (sn == 0 ? acc00 : acc01)
                               : (sn == 0 ? acc10 : acc11);
      if (mode != 2) {
#pragma unroll
        for (int r = 0; r < 8; ++r) {
          int row = mt * 32 + sm * 16 + r + 8 * half;
          int col = nt * 32 + sn * 16 + l16;
          Cout[(size_t)row * N + col] = (__bf16)(acc[r] * scale);
        }
      } else {
#pragma unroll
        for (int r = 0; r < 8; ++r) {
          int row = mt * 32 + sm * 16 + r + 8 * half;   // = b*S + s
          int col = nt * 32 + sn * 16 + l16;            // = h*64 + nn
          int bb = row >> 11, ss = row & 2047;
          int hh = col >> 6,  nn = col & 63;
          Cout[(size_t)(((bb * Hc + hh) << 6) + nn) * Sc + ss] = (__bf16)acc[r];
        }
      }
    }
  }
}

// scores + softmax: one block per (qtile, h, b). 8 waves build the 16x2048
// score strip in LDS via WMMA, then softmax and write fp32 probs to d_out.
static constexpr int SC_PITCH = Sc + 4;   // 2052: 8*2052 % 64 = 32 (bank skew)
__global__ __launch_bounds__(256)
void attn_scores_kernel(const __bf16* __restrict__ Qh,
                        const __bf16* __restrict__ Kh,
                        float* __restrict__ attnOut) {
  extern __shared__ float sc[];           // [16][SC_PITCH]
  int qt = blockIdx.x, h = blockIdx.y, b = blockIdx.z;
  int wv = threadIdx.x >> 5;
  int lane = threadIdx.x & 31, half = lane >> 4, l16 = lane & 15;

  const __bf16* qb =
      Qh + (size_t)(b * Sc + qt * 16 + l16) * HDc + h * DKc + 8 * half;
  v16bf a0 = load_frag_bf16(qb);
  v16bf a1 = load_frag_bf16(qb + 32);

  for (int nt = wv; nt < Sc / 16; nt += 8) {
    const __bf16* kb =
        Kh + (size_t)(b * Sc + nt * 16 + l16) * HDc + h * DKc + 8 * half;
    v8f acc = {};
    acc = wmma_bf16(a0, load_frag_bf16(kb), acc);
    acc = wmma_bf16(a1, load_frag_bf16(kb + 32), acc);
#pragma unroll
    for (int r = 0; r < 8; ++r)
      sc[(r + 8 * half) * SC_PITCH + nt * 16 + l16] = acc[r];
  }
  __syncthreads();

  // softmax: wave wv owns rows 2*wv and 2*wv+1
  for (int rr = 0; rr < 2; ++rr) {
    int row = wv * 2 + rr;
    float mx = -3.0e38f;
    for (int c = lane; c < Sc; c += 32)
      mx = fmaxf(mx, sc[row * SC_PITCH + c]);
#pragma unroll
    for (int o = 16; o > 0; o >>= 1) mx = fmaxf(mx, __shfl_xor(mx, o, 32));
    float sum = 0.f;
    for (int c = lane; c < Sc; c += 32) {
      float e = __expf(sc[row * SC_PITCH + c] - mx);
      sc[row * SC_PITCH + c] = e;
      sum += e;
    }
#pragma unroll
    for (int o = 16; o > 0; o >>= 1) sum += __shfl_xor(sum, o, 32);
    float inv = 1.0f / sum;
    size_t rbase = ((size_t)(h * Bc + b) * Sc + qt * 16 + row) * Sc;
    for (int c = lane; c < Sc; c += 32)
      attnOut[rbase + c] = sc[row * SC_PITCH + c] * inv;
  }
}

// context: O[b,q,h*64+n] = attn[h*B+b][q][:] @ Vt[b][h][n][:]
// Each wave owns 16 query rows and all 4 N-tiles (A-fragment converted once
// per K-step, reused 4x). Block = 8 waves = 128 rows. Output O stored bf16.
__global__ __launch_bounds__(256)
void attn_context_kernel(const float* __restrict__ attn,
                         const __bf16* __restrict__ Vt,
                         __bf16* __restrict__ O) {
  int mtile = blockIdx.x, h = blockIdx.y, b = blockIdx.z;
  int wv = threadIdx.x >> 5;
  int lane = threadIdx.x & 31, half = lane >> 4, l16 = lane & 15;
  int row = mtile * 128 + wv * 16 + l16;

  const float* arow =
      attn + ((size_t)(h * Bc + b) * Sc + row) * Sc + 8 * half;
  const __bf16* vbase =
      Vt + (size_t)((b * Hc + h) * DKc + l16) * Sc + 8 * half;

  v8f acc[4] = {};
  for (int k0 = 0; k0 < Sc; k0 += 32) {
    __builtin_prefetch(arow + k0 + 512, 0, 1);
    v16bf a = load_frag_f32(arow + k0);
#pragma unroll
    for (int ni = 0; ni < 4; ++ni) {
      v16bf bf = load_frag_bf16(vbase + (size_t)(ni * 16) * Sc + k0);
      acc[ni] = wmma_bf16(a, bf, acc[ni]);
    }
  }
#pragma unroll
  for (int ni = 0; ni < 4; ++ni) {
#pragma unroll
    for (int r = 0; r < 8; ++r) {
      int orow = mtile * 128 + wv * 16 + r + 8 * half;
      int ocol = h * DKc + ni * 16 + l16;
      O[(size_t)(b * Sc + orow) * HDc + ocol] = (__bf16)acc[ni][r];
    }
  }
}

// x = O(bf16) @ Wfc + bfc + residual (fp32 out to scratch); 2x2 blocked
__global__ __launch_bounds__(256)
void fc_gemm_kernel(const __bf16* __restrict__ O,
                    const __bf16* __restrict__ WtFc,
                    const float* __restrict__ bfc,
                    const float* __restrict__ residual,
                    float* __restrict__ X) {
  constexpr int N = Dc, K = HDc;
  int wave = blockIdx.x * 8 + (threadIdx.x >> 5);
  int lane = threadIdx.x & 31, half = lane >> 4, l16 = lane & 15;
  int mt = wave >> 5, nt = wave & 31;
  if (mt >= (BSc >> 5)) return;

  const __bf16* arow0 = O    + (size_t)(mt * 32 + l16) * K + 8 * half;
  const __bf16* arow1 = arow0 + (size_t)16 * K;
  const __bf16* brow0 = WtFc + (size_t)(nt * 32 + l16) * K + 8 * half;
  const __bf16* brow1 = brow0 + (size_t)16 * K;

  v8f acc00 = {}, acc01 = {}, acc10 = {}, acc11 = {};
  for (int k0 = 0; k0 < K; k0 += 32) {
    __builtin_prefetch(arow0 + k0 + 512, 0, 1);
    __builtin_prefetch(brow0 + k0 + 512, 0, 1);
    v16bf a0 = load_frag_bf16(arow0 + k0);
    v16bf a1 = load_frag_bf16(arow1 + k0);
    v16bf b0 = load_frag_bf16(brow0 + k0);
    v16bf b1 = load_frag_bf16(brow1 + k0);
    acc00 = wmma_bf16(a0, b0, acc00);
    acc01 = wmma_bf16(a0, b1, acc01);
    acc10 = wmma_bf16(a1, b0, acc10);
    acc11 = wmma_bf16(a1, b1, acc11);
  }

#pragma unroll
  for (int sm = 0; sm < 2; ++sm) {
#pragma unroll
    for (int sn = 0; sn < 2; ++sn) {
      const v8f& acc = sm == 0 ? (sn == 0 ? acc00 : acc01)
                               : (sn == 0 ? acc10 : acc11);
#pragma unroll
      for (int r = 0; r < 8; ++r) {
        int row = mt * 32 + sm * 16 + r + 8 * half;
        int col = nt * 32 + sn * 16 + l16;
        size_t idx = (size_t)row * N + col;
        X[idx] = acc[r] + bfc[col] + residual[idx];
      }
    }
  }
}

// row LayerNorm over D=1024; one block (8 waves) per row
__global__ __launch_bounds__(256)
void layernorm_kernel(const float* __restrict__ X,
                      const float* __restrict__ gamma,
                      const float* __restrict__ beta,
                      float* __restrict__ out) {
  __shared__ float red[2][8];
  int row = blockIdx.x, t = threadIdx.x;
  const float* xr = X + (size_t)row * Dc;
  float v[4], s = 0.f, s2 = 0.f;
#pragma unroll
  for (int i = 0; i < 4; ++i) {
    v[i] = xr[t + i * 256];
    s += v[i]; s2 += v[i] * v[i];
  }
#pragma unroll
  for (int o = 16; o > 0; o >>= 1) {
    s += __shfl_xor(s, o, 32);
    s2 += __shfl_xor(s2, o, 32);
  }
  int wv = t >> 5, lane = t & 31;
  if (lane == 0) { red[0][wv] = s; red[1][wv] = s2; }
  __syncthreads();
  if (wv == 0) {
    float a = (lane < 8) ? red[0][lane] : 0.f;
    float b = (lane < 8) ? red[1][lane] : 0.f;
#pragma unroll
    for (int o = 4; o > 0; o >>= 1) {
      a += __shfl_xor(a, o, 32);
      b += __shfl_xor(b, o, 32);
    }
    if (lane == 0) { red[0][0] = a; red[1][0] = b; }
  }
  __syncthreads();
  float mu   = red[0][0] * (1.0f / Dc);
  float var  = red[1][0] * (1.0f / Dc) - mu * mu;
  float rstd = rsqrtf(var + 1e-5f);
#pragma unroll
  for (int i = 0; i < 4; ++i) {
    int c = t + i * 256;
    out[(size_t)row * Dc + c] = (v[i] - mu) * rstd * gamma[c] + beta[c];
  }
}

// ---------------------------------------------------------------- launch
extern "C" void kernel_launch(void* const* d_in, const int* in_sizes, int n_in,
                              void* d_out, int out_size, void* d_ws, size_t ws_size,
                              hipStream_t stream) {
  const float* q     = (const float*)d_in[0];
  const float* k     = (const float*)d_in[1];
  const float* v     = (const float*)d_in[2];
  const float* Wq    = (const float*)d_in[3];
  const float* Wk    = (const float*)d_in[4];
  const float* Wv    = (const float*)d_in[5];
  const float* Wfc   = (const float*)d_in[6];
  const float* bfc   = (const float*)d_in[7];
  const float* gamma = (const float*)d_in[8];
  const float* beta  = (const float*)d_in[9];

  constexpr size_t MB = 1024ull * 1024ull;
  uint8_t* ws = (uint8_t*)d_ws;
  __bf16* WT_Q  = (__bf16*)(ws + 0 * MB);    // 2 MB each
  __bf16* WT_K  = (__bf16*)(ws + 2 * MB);
  __bf16* WT_V  = (__bf16*)(ws + 4 * MB);
  __bf16* WT_FC = (__bf16*)(ws + 6 * MB);
  __bf16* QB    = (__bf16*)(ws + 8 * MB);    // 16 MB: q in bf16
  __bf16* KB    = (__bf16*)(ws + 24 * MB);   // 16 MB: k in bf16
  __bf16* VB    = (__bf16*)(ws + 40 * MB);   // 16 MB: v in bf16
  __bf16* QH    = (__bf16*)(ws + 56 * MB);   // 16 MB
  __bf16* KH    = (__bf16*)(ws + 72 * MB);   // 16 MB
  __bf16* VT    = (__bf16*)(ws + 88 * MB);   // 16 MB
  __bf16* OB    = (__bf16*)(ws + 104 * MB);  // 16 MB: context out, bf16
  float*  Xbuf  = (float*)(ws + 8 * MB);     // 32 MB: reuses QB+KB (dead)

  float* outLN = (float*)d_out;
  float* attn  = (float*)d_out + (size_t)Bc * Sc * Dc;      // + 8,388,608

  // 1) weight transpose + bf16 convert; q/k/v bulk bf16 convert
  transpose_w_kernel<<<4096, 256, 0, stream>>>(Wq,  WT_Q);
  transpose_w_kernel<<<4096, 256, 0, stream>>>(Wk,  WT_K);
  transpose_w_kernel<<<4096, 256, 0, stream>>>(Wv,  WT_V);
  transpose_w_kernel<<<4096, 256, 0, stream>>>(Wfc, WT_FC);
  cvt_bf16_kernel<<<4096, 256, 0, stream>>>(q, QB);   // 8M elems / 8 per thread
  cvt_bf16_kernel<<<4096, 256, 0, stream>>>(k, KB);
  cvt_bf16_kernel<<<4096, 256, 0, stream>>>(v, VB);

  // 2) projections (Q scaled by 1/sqrt(DK); V stored per-head transposed)
  proj_gemm_kernel<<<1024, 256, 0, stream>>>(QB, WT_Q, QH, 0.125f, 0);
  proj_gemm_kernel<<<1024, 256, 0, stream>>>(KB, WT_K, KH, 1.0f,   0);
  proj_gemm_kernel<<<1024, 256, 0, stream>>>(VB, WT_V, VT, 1.0f,   2);

  // 3) scores + softmax -> fp32 probs in d_out
  size_t smem = (size_t)16 * SC_PITCH * sizeof(float);      // ~128 KB (< 320 KB/WGP)
  attn_scores_kernel<<<dim3(Sc / 16, Hc, Bc), 256, smem, stream>>>(QH, KH, attn);

  // 4) attn @ V -> O (bf16)
  attn_context_kernel<<<dim3(Sc / 128, Hc, Bc), 256, 0, stream>>>(attn, VT, OB);

  // 5) FC + bias + residual
  fc_gemm_kernel<<<1024, 256, 0, stream>>>(OB, WT_FC, bfc, q, Xbuf);

  // 6) LayerNorm -> final out
  layernorm_kernel<<<BSc, 256, 0, stream>>>(Xbuf, gamma, beta, outLN);
}
